// Model_1967095021831
// MI455X (gfx1250) — compile-verified, matrix operand order checked by reading, not run on hardware
//
#include <hip/hip_runtime.h>
#include <hip/hip_bf16.h>
#include <math.h>

// ---- problem constants (match reference) ----
#define B_   16
#define L_   720
#define C_   862
#define H_   512
#define P_   96
#define K_MA 25
#define PAD_MA 12
#define PERIOD_ 24
#define NSTEP_  10
#define DT_  0.1f
#define KIN_ 0.1f
#define JAC_ 0.1f
#define EPS_ 1e-5f
#define M_   (B_ * C_)   // 13792 rows for all GEMMs

// ---- WMMA types ----
typedef __attribute__((ext_vector_type(16))) __bf16 v16bf;
typedef __attribute__((ext_vector_type(2)))  __bf16 v2bf;
typedef __attribute__((ext_vector_type(2)))  float  v2f;
typedef __attribute__((ext_vector_type(8)))  float  v8f;

// packed fp32->bf16 conversion: single v_cvt_pk_bf16_f32 per pair
__device__ __forceinline__ unsigned int pk_bf16(float x, float y) {
  v2f f; f.x = x; f.y = y;
  union { v2bf v; unsigned int u; } r;
  r.v = __builtin_convertvector(f, v2bf);
  return r.u;
}
__device__ __forceinline__ unsigned short f2bf(float f) {
  union { v2bf v; unsigned short s[2]; } r;
  v2f t; t.x = f; t.y = 0.f;
  r.v = __builtin_convertvector(t, v2bf);
  return r.s[0];
}

// =====================================================================
// WMMA GEMM: out[M,N] = epilogue( A[M,K] @ W[N,K]^T + bias )
//   A fp32 (v_cvt_pk_bf16_f32 while staging to LDS), W bf16 (pre-converted,
//   staged via global_load_async_to_lds_b128 on full tiles)
//   mode 0: out = acc + bias
//   mode 1: out = tanh(acc + bias)
//   mode 2: out = (1 - aux^2) * acc        (JVP tangent through tanh)
// Tile: BM=128, BN=128, BK=32; 256 threads = 8 wave32, each wave 32x64.
// =====================================================================
#define BM 128
#define BN 128
#define BK 32

__global__ __launch_bounds__(256)
void gemm_bf16_kernel(const float* __restrict__ A,
                      const unsigned short* __restrict__ Wb,
                      const float* __restrict__ bias,
                      const float* __restrict__ aux,
                      float* __restrict__ out,
                      int M, int N, int K, int mode)
{
  __shared__ unsigned short As[BM * BK];   // 8 KB
  __shared__ unsigned short Bs[BN * BK];   // 8 KB

  const int tid  = threadIdx.x;
  const int m0   = blockIdx.x * BM;
  const int n0   = blockIdx.y * BN;
  const int lane = tid & 31;
  const int wave = tid >> 5;
  const int wm   = (wave & 3) * 32;   // 4 waves tile M (4*32 = 128)
  const int wn   = (wave >> 2) * 64;  // 2 waves tile N (2*64 = 128)
  const int lidx = lane & 15;
  const int grp  = (lane >> 4) & 1;
  const int kbase = grp * 8;          // 16-bit A/B layout: lane group selects K-half

  const bool mFull = (m0 + BM <= M);
  const bool nFull = (n0 + BN <= N);

  v8f acc[2][4] = {};

  // A staging addresses (4 float4 per thread)
  const int aRow[4] = { (tid + 0*256) >> 3, (tid + 1*256) >> 3,
                        (tid + 2*256) >> 3, (tid + 3*256) >> 3 };
  const int aCol = (tid & 7) * 4;
  // B staging addresses (2 x 16B per thread)
  const int bRow[2] = { (tid + 0*256) >> 2, (tid + 1*256) >> 2 };
  const int bCol = (tid & 3) * 8;
  // per-lane LDS byte offsets of the B destinations (low 32 bits of LDS addr)
  unsigned bLds[2];
  bLds[0] = (unsigned)(size_t)(&Bs[bRow[0] * BK + bCol]);
  bLds[1] = (unsigned)(size_t)(&Bs[bRow[1] * BK + bCol]);

  const int nK = (K + BK - 1) / BK;
  for (int kt = 0; kt < nK; ++kt) {
    const int k0 = kt * BK;
    const bool kFull = (k0 + BK <= K);

    // ---- stage B tile (128x32 bf16) ----
    bool asyncB = nFull && kFull;
    if (asyncB) {
      // CDNA5 async copy: global -> LDS without VGPR round-trip (ASYNCcnt)
#pragma unroll
      for (int i = 0; i < 2; ++i) {
        const unsigned short* gp = Wb + (size_t)(n0 + bRow[i]) * K + (k0 + bCol);
        asm volatile("global_load_async_to_lds_b128 %0, %1, off"
                     :: "v"(bLds[i]), "v"(gp) : "memory");
      }
    } else {
#pragma unroll
      for (int i = 0; i < 2; ++i) {
        int gn = n0 + bRow[i], gk = k0 + bCol;
        uint4 v = make_uint4(0u, 0u, 0u, 0u);
        if (gn < N && gk < K) v = *(const uint4*)(Wb + (size_t)gn * K + gk);
        *(uint4*)(&Bs[bRow[i] * BK + bCol]) = v;
      }
    }

    // ---- stage A tile (128x32 fp32 -> bf16 in LDS) ----
    if (mFull && kFull) {               // uniform fast path, no exec juggling
#pragma unroll
      for (int i = 0; i < 4; ++i) {
        float4 v = *(const float4*)(A + (size_t)(m0 + aRow[i]) * K + (k0 + aCol));
        *(uint2*)(&As[aRow[i] * BK + aCol]) =
            make_uint2(pk_bf16(v.x, v.y), pk_bf16(v.z, v.w));
      }
    } else {
#pragma unroll
      for (int i = 0; i < 4; ++i) {
        int gm = m0 + aRow[i], gk = k0 + aCol;
        float4 v = make_float4(0.f, 0.f, 0.f, 0.f);
        if (gm < M && gk < K) v = *(const float4*)(A + (size_t)gm * K + gk);
        *(uint2*)(&As[aRow[i] * BK + aCol]) =
            make_uint2(pk_bf16(v.x, v.y), pk_bf16(v.z, v.w));
      }
    }

    if (asyncB) {
      asm volatile("s_wait_asynccnt 0x0" ::: "memory");   // B tile resident in LDS
    }
    __syncthreads();

    // ---- fragments (2 x ds_load_b128 each, per documented 16-bit layout) ----
    union Frag { v16bf v; uint4 q[2]; };
    Frag a[2], b[4];
#pragma unroll
    for (int mi = 0; mi < 2; ++mi) {
      int r = wm + mi * 16 + lidx;
      a[mi].q[0] = *(const uint4*)(&As[r * BK + kbase]);
      a[mi].q[1] = *(const uint4*)(&As[r * BK + kbase + 16]);
    }
#pragma unroll
    for (int ni = 0; ni < 4; ++ni) {
      int r = wn + ni * 16 + lidx;
      b[ni].q[0] = *(const uint4*)(&Bs[r * BK + kbase]);
      b[ni].q[1] = *(const uint4*)(&Bs[r * BK + kbase + 16]);
    }
#pragma unroll
    for (int mi = 0; mi < 2; ++mi)
#pragma unroll
      for (int ni = 0; ni < 4; ++ni)
        acc[mi][ni] = __builtin_amdgcn_wmma_f32_16x16x32_bf16(
            false, a[mi].v, false, b[ni].v, (short)0, acc[mi][ni], false, false);
    __syncthreads();
  }

  // ---- epilogue ----
#pragma unroll
  for (int mi = 0; mi < 2; ++mi) {
#pragma unroll
    for (int ni = 0; ni < 4; ++ni) {
      int gn = n0 + wn + ni * 16 + lidx;
      float bv = 0.f;
      if (mode != 2 && bias != nullptr && gn < N) bv = bias[gn];
      if (mFull && nFull) {
#pragma unroll
        for (int r = 0; r < 8; ++r) {
          int gm = m0 + wm + mi * 16 + grp * 8 + r;
          float v = acc[mi][ni][r];
          if (mode == 0)      v = v + bv;
          else if (mode == 1) v = tanhf(v + bv);
          else { float h = aux[(size_t)gm * N + gn]; v = (1.f - h * h) * v; }
          out[(size_t)gm * N + gn] = v;
        }
      } else {
#pragma unroll
        for (int r = 0; r < 8; ++r) {
          int gm = m0 + wm + mi * 16 + grp * 8 + r;
          if (gm < M && gn < N) {
            float v = acc[mi][ni][r];
            if (mode == 0)      v = v + bv;
            else if (mode == 1) v = tanhf(v + bv);
            else { float h = aux[(size_t)gm * N + gn]; v = (1.f - h * h) * v; }
            out[(size_t)gm * N + gn] = v;
          }
        }
      }
    }
  }
}

// =====================================================================
// elementwise / reduction kernels
// =====================================================================
__global__ void cvt_bf16_kernel(const float* __restrict__ in,
                                unsigned short* __restrict__ out, int n) {
  int i = blockIdx.x * 256 + threadIdx.x;
  if (i < n) out[i] = f2bf(in[i]);
}

// moving average (edge-replicated, K=25) + residual, written transposed [B*C, L]
__global__ void decomp_kernel(const float* __restrict__ x,
                              float* __restrict__ m_t, float* __restrict__ res_t) {
  int bc = blockIdx.x * blockDim.x + threadIdx.x;
  if (bc >= M_) return;
  int b = bc / C_, c = bc % C_;
  const float* xp = x + (size_t)b * L_ * C_ + c;   // element l at xp[l*C_]
  float sum = 0.f;
  for (int i = -PAD_MA; i <= PAD_MA; ++i) {
    int j = i < 0 ? 0 : (i > L_ - 1 ? L_ - 1 : i);
    sum += xp[(size_t)j * C_];
  }
  float* mrow = m_t  + (size_t)bc * L_;
  float* rrow = res_t + (size_t)bc * L_;
  for (int l = 0; l < L_; ++l) {
    float m  = sum * (1.0f / (float)K_MA);
    float xv = xp[(size_t)l * C_];
    mrow[l] = m;
    rrow[l] = xv - m;
    int add = l + PAD_MA + 1; if (add > L_ - 1) add = L_ - 1;
    int sub = l - PAD_MA;     if (sub < 0)      sub = 0;
    sum += xp[(size_t)add * C_] - xp[(size_t)sub * C_];
  }
}

// per-phase seasonal means, centered over PERIOD
__global__ void seas_kernel(const float* __restrict__ res_t, float* __restrict__ seas) {
  int bc = blockIdx.x;
  int p  = threadIdx.x;
  __shared__ float sbuf[33];
  float s = 0.f;
  if (p < PERIOD_) {
    const float* r = res_t + (size_t)bc * L_;
    for (int i = 0; i < L_ / PERIOD_; ++i) s += r[i * PERIOD_ + p];
    s *= (float)PERIOD_ / (float)L_;    // / (L/PERIOD)
  }
  sbuf[p] = (p < PERIOD_) ? s : 0.f;
  __syncthreads();
  if (p == 0) {
    float tot = 0.f;
    for (int i = 0; i < PERIOD_; ++i) tot += sbuf[i];
    sbuf[32] = tot / (float)PERIOD_;
  }
  __syncthreads();
  if (p < PERIOD_) seas[(size_t)bc * PERIOD_ + p] = s - sbuf[32];
}

// RevIN normalize + implicit [B,L,C]->[B*C,L] transpose; branch selects series
__global__ void norm_kernel(const float* __restrict__ m_t, const float* __restrict__ res_t,
                            const float* __restrict__ seas,
                            const float* __restrict__ rw, const float* __restrict__ rb,
                            float* __restrict__ nX, float* __restrict__ mu_buf,
                            float* __restrict__ sd_buf, int branch) {
  int bc = blockIdx.x * blockDim.x + threadIdx.x;
  if (bc >= M_) return;
  int c = bc % C_;
  const float* mrow = m_t  + (size_t)bc * L_;
  const float* rrow = res_t + (size_t)bc * L_;
  const float* srow = seas  + (size_t)bc * PERIOD_;
  float sum = 0.f, sq = 0.f;
  for (int l = 0; l < L_; ++l) {
    float v;
    if (branch == 0)      v = mrow[l];
    else if (branch == 1) v = srow[l % PERIOD_];
    else                  v = rrow[l] - srow[l % PERIOD_];
    sum += v; sq += v * v;
  }
  float mu  = sum / (float)L_;
  float var = sq / (float)L_ - mu * mu;
  if (var < 0.f) var = 0.f;
  float sd  = sqrtf(var + EPS_);
  float inv = 1.0f / sd;
  float w = rw[c], b0 = rb[c];
  float* orow = nX + (size_t)bc * L_;
  for (int l = 0; l < L_; ++l) {
    float v;
    if (branch == 0)      v = mrow[l];
    else if (branch == 1) v = srow[l % PERIOD_];
    else                  v = rrow[l] - srow[l % PERIOD_];
    orow[l] = (v - mu) * inv * w + b0;
  }
  mu_buf[bc] = mu; sd_buf[bc] = sd;
}

// eps [B,H,C] -> eps_t [B*C,H]
__global__ void transpose_eps_kernel(const float* __restrict__ eps,
                                     float* __restrict__ eps_t) {
  int idx = blockIdx.x * 256 + threadIdx.x;
  if (idx >= M_ * H_) return;
  int bc = idx / H_, h = idx % H_;
  int b = bc / C_, c = bc % C_;
  eps_t[idx] = eps[((size_t)b * H_ + h) * C_ + c];
}

// Euler step + kinetic/jacobian regularizer partial sums (grid divides exactly)
__global__ __launch_bounds__(256)
void ode_update_kernel(float* __restrict__ z, const float* __restrict__ dz,
                       const float* __restrict__ jeps,
                       float* __restrict__ kvec, float* __restrict__ jvec) {
  int idx = blockIdx.x * 256 + threadIdx.x;     // M_*H_ divisible by 256
  float d  = dz[idx];
  float je = jeps[idx];
  z[idx] += DT_ * d;
  __shared__ float s1[256], s2[256];
  s1[threadIdx.x] = d * d;
  s2[threadIdx.x] = je * je;
  __syncthreads();
  for (int o = 128; o > 0; o >>= 1) {
    if (threadIdx.x < o) { s1[threadIdx.x] += s1[threadIdx.x + o];
                           s2[threadIdx.x] += s2[threadIdx.x + o]; }
    __syncthreads();
  }
  if (threadIdx.x == 0) {
    int b = idx / (C_ * H_);                    // C_*H_ divisible by 256
    const float scale = DT_ / (float)(C_ * H_);
    atomicAdd(&kvec[b], s1[0] * scale);
    atomicAdd(&jvec[b], s2[0] * scale);
  }
}

__global__ void sde_tmp_kernel(const float* __restrict__ r, const float* __restrict__ g1,
                               const float* __restrict__ dW, int step,
                               float* __restrict__ out) {
  int idx = blockIdx.x * 256 + threadIdx.x;
  if (idx >= M_ * H_) return;
  int b = idx / (C_ * H_);
  float dw = dW[step * B_ + b];
  out[idx] = r[idx] + g1[idx] * dw;
}

__global__ void sde_update_kernel(float* __restrict__ r, const float* __restrict__ fy,
                                  const float* __restrict__ g1, const float* __restrict__ g2,
                                  const float* __restrict__ dW, int step) {
  int idx = blockIdx.x * 256 + threadIdx.x;
  if (idx >= M_ * H_) return;
  int b = idx / (C_ * H_);
  float dw = dW[step * B_ + b];
  r[idx] += DT_ * fy[idx] + 0.5f * (g1[idx] + g2[idx]) * dw;
}

// denorm + transpose [B*C,P] -> [B,P,C], accumulate into output
__global__ void dec_acc_kernel(const float* __restrict__ yb,
                               const float* __restrict__ rw, const float* __restrict__ rb,
                               const float* __restrict__ mu_buf, const float* __restrict__ sd_buf,
                               float* __restrict__ out) {
  int idx = blockIdx.x * 256 + threadIdx.x;
  if (idx >= M_ * P_) return;
  int bc = idx / P_, p = idx % P_;
  int b = bc / C_, c = bc % C_;
  float y = yb[idx];
  y = (y - rb[c]) / rw[c] * sd_buf[bc] + mu_buf[bc];
  out[((size_t)b * P_ + p) * C_ + c] += y;
}

__global__ void reg_final_kernel(const float* __restrict__ kj, float* __restrict__ out_scalar) {
  if (blockIdx.x == 0 && threadIdx.x == 0) {
    float r = 0.f;
    for (int b = 0; b < B_; ++b)
      r += KIN_ * kj[b] + JAC_ * kj[B_ + b] + KIN_ * kj[2 * B_ + b] + JAC_ * kj[3 * B_ + b];
    out_scalar[0] = r / (float)B_;
  }
}

// =====================================================================
// host orchestration
// =====================================================================
extern "C" void kernel_launch(void* const* d_in, const int* in_sizes, int n_in,
                              void* d_out, int out_size, void* d_ws, size_t ws_size,
                              hipStream_t stream) {
  (void)in_sizes; (void)n_in; (void)ws_size;
  const float* x     = (const float*)d_in[0];
  // d_in[1] = t, unused
  const float* dWn   = (const float*)d_in[2];   // [NSTEP_, B_]
  const float* eps_T = (const float*)d_in[3];
  const float* eps_S = (const float*)d_in[4];
  const float* pp[30];
  for (int i = 0; i < 30; ++i) pp[i] = (const float*)d_in[5 + i];
  // pp layout: s in {T,S,R}: [s*6+0]=enc_W [s*6+1]=enc_b [s*6+2]=dec_W [s*6+3]=dec_b
  //                          [s*6+4]=revin_w [s*6+5]=revin_b
  // [18..21]=ode_T W1,b1,W2,b2  [22..25]=ode_S  [26..29]=sde muW,mub,sgW,sgb

  char* base = (char*)d_ws;
  size_t off = 0;
  auto alloc = [&](size_t bytes) -> char* {
    off = (off + 255) & ~(size_t)255;
    char* p = base + off;
    off += bytes;
    return p;
  };
  float* m_t   = (float*)alloc((size_t)M_ * L_ * 4);
  float* res_t = (float*)alloc((size_t)M_ * L_ * 4);
  float* seas  = (float*)alloc((size_t)M_ * PERIOD_ * 4);
  float* nX    = (float*)alloc((size_t)M_ * L_ * 4);
  float* mu_b  = (float*)alloc((size_t)M_ * 4);
  float* sd_b  = (float*)alloc((size_t)M_ * 4);
  float* eps_t = (float*)alloc((size_t)M_ * H_ * 4);
  float* z     = (float*)alloc((size_t)M_ * H_ * 4);
  float* buf1  = (float*)alloc((size_t)M_ * H_ * 4);
  float* buf2  = (float*)alloc((size_t)M_ * H_ * 4);
  float* buf3  = (float*)alloc((size_t)M_ * H_ * 4);
  float* buf4  = (float*)alloc((size_t)M_ * H_ * 4);
  float* ybuf  = (float*)alloc((size_t)M_ * P_ * 4);
  float* kj    = (float*)alloc((size_t)4 * B_ * 4);
  unsigned short* encWb[3], *decWb[3], *odeWb[4], *sdeWb[2];
  for (int s = 0; s < 3; ++s) encWb[s] = (unsigned short*)alloc((size_t)H_ * L_ * 2);
  for (int s = 0; s < 3; ++s) decWb[s] = (unsigned short*)alloc((size_t)P_ * H_ * 2);
  for (int s = 0; s < 4; ++s) odeWb[s] = (unsigned short*)alloc((size_t)H_ * H_ * 2);
  for (int s = 0; s < 2; ++s) sdeWb[s] = (unsigned short*)alloc((size_t)H_ * H_ * 2);

  hipMemsetAsync(kj, 0, (size_t)4 * B_ * 4, stream);
  hipMemsetAsync(d_out, 0, (size_t)out_size * 4, stream);

  auto cvt = [&](const float* in, unsigned short* out, int n) {
    cvt_bf16_kernel<<<(n + 255) / 256, 256, 0, stream>>>(in, out, n);
  };
  for (int s = 0; s < 3; ++s) {
    cvt(pp[s * 6 + 0], encWb[s], H_ * L_);
    cvt(pp[s * 6 + 2], decWb[s], P_ * H_);
  }
  cvt(pp[18], odeWb[0], H_ * H_);  cvt(pp[20], odeWb[1], H_ * H_);  // T: W1, W2
  cvt(pp[22], odeWb[2], H_ * H_);  cvt(pp[24], odeWb[3], H_ * H_);  // S: W1, W2
  cvt(pp[26], sdeWb[0], H_ * H_);  cvt(pp[28], sdeWb[1], H_ * H_);  // mu, sigma

  auto gemm = [&](const float* A, const unsigned short* Wb, const float* bias,
                  const float* aux, float* out, int M, int N, int K, int mode) {
    dim3 g((M + BM - 1) / BM, (N + BN - 1) / BN);
    gemm_bf16_kernel<<<g, dim3(256), 0, stream>>>(A, Wb, bias, aux, out, M, N, K, mode);
  };

  decomp_kernel<<<(M_ + 255) / 256, 256, 0, stream>>>(x, m_t, res_t);
  seas_kernel<<<M_, 32, 0, stream>>>(res_t, seas);

  const int nMH = (M_ * H_) / 256;          // exact
  for (int s = 0; s < 3; ++s) {
    const float* rw = pp[s * 6 + 4];
    const float* rb = pp[s * 6 + 5];
    norm_kernel<<<(M_ + 255) / 256, 256, 0, stream>>>(m_t, res_t, seas, rw, rb,
                                                      nX, mu_b, sd_b, s);
    gemm(nX, encWb[s], pp[s * 6 + 1], nullptr, z, M_, H_, L_, /*tanh*/1);

    if (s < 2) {  // ODE branches (trend, seasonal)
      const float* eps = (s == 0) ? eps_T : eps_S;
      transpose_eps_kernel<<<(M_ * H_ + 255) / 256, 256, 0, stream>>>(eps, eps_t);
      const unsigned short* W1 = odeWb[s * 2 + 0];
      const unsigned short* W2 = odeWb[s * 2 + 1];
      const float* b1 = pp[18 + s * 4 + 1];
      const float* b2 = pp[18 + s * 4 + 3];
      float* kvec = kj + (2 * s) * B_;
      float* jvec = kj + (2 * s + 1) * B_;
      for (int st = 0; st < NSTEP_; ++st) {
        gemm(z,     W1, b1,      nullptr, buf1, M_, H_, H_, 1);  // h1 = tanh(z W1^T + b1)
        gemm(eps_t, W1, nullptr, buf1,    buf2, M_, H_, H_, 2);  // th1 = (1-h1^2)(eps W1^T)
        gemm(buf1,  W2, b2,      nullptr, buf3, M_, H_, H_, 1);  // dz = h2 = tanh(h1 W2^T + b2)
        gemm(buf2,  W2, nullptr, buf3,    buf4, M_, H_, H_, 2);  // Jeps = (1-h2^2)(th1 W2^T)
        ode_update_kernel<<<nMH, 256, 0, stream>>>(z, buf3, buf4, kvec, jvec);
      }
    } else {      // SDE branch (residual), Stratonovich Euler-Heun
      const float* bmu = pp[27];
      const float* bsg = pp[29];
      for (int st = 0; st < NSTEP_; ++st) {
        gemm(z, sdeWb[1], bsg, nullptr, buf1, M_, H_, H_, 0);    // g1
        gemm(z, sdeWb[0], bmu, nullptr, buf2, M_, H_, H_, 0);    // fy
        sde_tmp_kernel<<<nMH, 256, 0, stream>>>(z, buf1, dWn, st, buf3);
        gemm(buf3, sdeWb[1], bsg, nullptr, buf4, M_, H_, H_, 0); // g2
        sde_update_kernel<<<nMH, 256, 0, stream>>>(z, buf2, buf1, buf4, dWn, st);
      }
    }

    gemm(z, decWb[s], pp[s * 6 + 3], nullptr, ybuf, M_, P_, H_, 0);
    dec_acc_kernel<<<(M_ * P_ + 255) / 256, 256, 0, stream>>>(ybuf, rw, rb, mu_b, sd_b,
                                                              (float*)d_out);
  }
  reg_final_kernel<<<1, 32, 0, stream>>>(kj, (float*)d_out + (size_t)B_ * P_ * C_);
}
